// EncoderBlock_14783277432890
// MI455X (gfx1250) — matrix operand order repaired and checked by reading, not compiled, not run on hardware
//
#include <hip/hip_runtime.h>
#include <stdint.h>
#include <stddef.h>

// ---------------- problem constants ----------------
constexpr int Bb  = 8;
constexpr int Ss  = 1024;
constexpr int Ee  = 1024;
constexpr int Hh  = 16;
constexpr int Dk  = 64;       // Ee / Hh
constexpr int Dff = 4096;
constexpr int Mm  = Bb * Ss;  // 8192 rows for every GEMM

// ---------------- types ----------------
typedef __bf16 bf16;
typedef __attribute__((ext_vector_type(8)))  __bf16 v8bf;
typedef __attribute__((ext_vector_type(16))) __bf16 v16bf;
typedef __attribute__((ext_vector_type(8)))  float  v8f;

static __device__ __forceinline__ bf16 f2bf(float f) {
    union { float f; uint32_t u; } x; x.f = f;
    uint32_t r = (x.u + 0x7FFFu + ((x.u >> 16) & 1u)) >> 16;  // RNE
    union { uint16_t u; bf16 b; } y; y.u = (uint16_t)r;
    return y.b;
}

static __device__ __forceinline__ v16bf cat16(v8bf a, v8bf b) {
    return __builtin_shufflevector(a, b, 0,1,2,3,4,5,6,7,8,9,10,11,12,13,14,15);
}

// async copy of 16 bytes global -> LDS (CDNA5 ASYNCcnt path, no VGPR round trip)
static __device__ __forceinline__ void async_b128(const void* gsrc, void* ldst) {
    unsigned           l = (unsigned)(size_t)ldst;            // LDS byte address
    unsigned long long g = (unsigned long long)(size_t)gsrc;  // global address
    asm volatile("global_load_async_to_lds_b128 %0, %1, off"
                 :: "v"(l), "v"(g) : "memory");
}
static __device__ __forceinline__ void wait_async0() {
    asm volatile("s_wait_asynccnt 0x0" ::: "memory");
}

// ---------------- f32 -> bf16 convert ----------------
__global__ void f32_to_bf16_kernel(const float* __restrict__ src,
                                   bf16* __restrict__ dst, int n) {
    int i = blockIdx.x * blockDim.x + threadIdx.x;
    if (i < n) dst[i] = f2bf(src[i]);
}

// ---------------- LayerNorm (unbiased var, /(std+eps)) ----------------
__global__ __launch_bounds__(256)
void layernorm_bf16_kernel(const float* __restrict__ x,
                           const float* __restrict__ alpha,
                           const float* __restrict__ beta,
                           bf16* __restrict__ out) {
    __shared__ float red[256];
    const int row = blockIdx.x;
    const int t   = threadIdx.x;
    const float* xr = x + (size_t)row * Ee;

    float s = 0.f;
#pragma unroll
    for (int k = 0; k < Ee / 256; ++k) s += xr[t + k * 256];
    red[t] = s; __syncthreads();
    for (int off = 128; off > 0; off >>= 1) {
        if (t < off) red[t] += red[t + off];
        __syncthreads();
    }
    const float mean = red[0] / (float)Ee;
    __syncthreads();

    float ss = 0.f;
#pragma unroll
    for (int k = 0; k < Ee / 256; ++k) {
        float d = xr[t + k * 256] - mean;
        ss += d * d;
    }
    red[t] = ss; __syncthreads();
    for (int off = 128; off > 0; off >>= 1) {
        if (t < off) red[t] += red[t + off];
        __syncthreads();
    }
    const float var = red[0] / (float)(Ee - 1);
    const float inv = 1.f / (sqrtf(var) + 1e-6f);

#pragma unroll
    for (int k = 0; k < Ee / 256; ++k) {
        int i = t + k * 256;
        out[(size_t)row * Ee + i] = f2bf(alpha[i] * (xr[i] - mean) * inv + beta[i]);
    }
}

// ---------------- bf16 WMMA GEMM: C = A(MxK) * W(KxN) + bias ----------------
// 128x128 block tile, K-step 32, 8 waves, each wave -> 64x32 (4x2 wmma tiles)
// Double-buffered LDS; A tile via GLOBAL_LOAD_ASYNC_TO_LDS_B128 overlapped with
// WMMA of the current tile; one barrier per K-step.
__global__ __launch_bounds__(256)
void gemm_bf16_kernel(const bf16* __restrict__ A, const bf16* __restrict__ W,
                      const float* __restrict__ bias,
                      float* outF, bf16* outB, const float* __restrict__ resid,
                      int M, int N, int K, int relu) {
    __shared__ bf16 As[2][128][32];   // row-major A tiles
    __shared__ bf16 Ws[2][128][32];   // W tiles stored transposed: [n][k]

    const int t    = threadIdx.x;
    const int lane = t & 31;
    const int wave = t >> 5;
    const int wr   = wave >> 2;    // 0..1
    const int wc   = wave & 3;     // 0..3
    const int br   = blockIdx.y * 128;
    const int bc   = blockIdx.x * 128;

    v8f acc[4][2];
    const v8f zf = {0.f,0.f,0.f,0.f,0.f,0.f,0.f,0.f};
#pragma unroll
    for (int i = 0; i < 4; ++i)
#pragma unroll
        for (int j = 0; j < 2; ++j) acc[i][j] = zf;

    const int rA = t >> 1, cA = (t & 1) * 16;   // A tile: 2 threads / row
    const int kW = t >> 3, nW = (t & 7) * 16;   // W tile: 8 threads / k-row
    const int mrow = lane & 15;
    const int g    = lane >> 4;

    auto issueA = [&](int buf, int kk) {
        const bf16* ga = &A[(size_t)(br + rA) * K + kk + cA];
        async_b128(ga,     &As[buf][rA][cA]);
        async_b128(ga + 8, &As[buf][rA][cA + 8]);
    };
    auto stageW = [&](int buf, v8bf w0, v8bf w1) {
#pragma unroll
        for (int u = 0; u < 8; ++u) {
            Ws[buf][nW + u][kW]     = w0[u];
            Ws[buf][nW + 8 + u][kW] = w1[u];
        }
    };

    // prologue: stage tile 0 (no one reads buffer 0 yet)
    issueA(0, 0);
    {
        const bf16* gw = &W[(size_t)kW * N + bc + nW];
        stageW(0, *(const v8bf*)gw, *(const v8bf*)(gw + 8));
    }

    const int nsteps = K >> 5;
    for (int s = 0; s < nsteps; ++s) {
        const int  cur     = s & 1;
        const int  nxt     = cur ^ 1;
        const bool hasNext = (s + 1 < nsteps);

        // hoist next W-tile global loads (latency hides under wait + barrier)
        v8bf w0 = {}, w1 = {};
        if (hasNext) {
            const bf16* gw = &W[(size_t)((s + 1) * 32 + kW) * N + bc + nW];
            w0 = *(const v8bf*)gw;
            w1 = *(const v8bf*)(gw + 8);
        }

        wait_async0();       // our async writes into As[cur] have landed
        __syncthreads();     // publish As[cur] + Ws[cur]; prev readers of nxt done

        if (hasNext) {       // overlap next-tile staging with current WMMAs
            issueA(nxt, (s + 1) * 32);
            stageW(nxt, w0, w1);
        }

        // Fragments per ISA 7.12.2 layouts
        v16bf a[4], b[2];
#pragma unroll
        for (int i = 0; i < 4; ++i) {
            int r = wr * 64 + i * 16 + mrow;
            a[i] = cat16(*(const v8bf*)&As[cur][r][g * 8],
                         *(const v8bf*)&As[cur][r][16 + g * 8]);
        }
#pragma unroll
        for (int j = 0; j < 2; ++j) {
            int n = wc * 32 + j * 16 + mrow;
            b[j] = cat16(*(const v8bf*)&Ws[cur][n][g * 8],
                         *(const v8bf*)&Ws[cur][n][16 + g * 8]);
        }
#pragma unroll
        for (int i = 0; i < 4; ++i)
#pragma unroll
            for (int j = 0; j < 2; ++j)
                acc[i][j] = __builtin_amdgcn_wmma_f32_16x16x32_bf16(
                    false, a[i], false, b[j], (short)0, acc[i][j], false, false);
    }

    // Epilogue: bias (+relu) (+residual), store f32 and/or bf16
#pragma unroll
    for (int i = 0; i < 4; ++i)
#pragma unroll
        for (int j = 0; j < 2; ++j)
#pragma unroll
            for (int e = 0; e < 8; ++e) {
                int row = br + wr * 64 + i * 16 + e + 8 * g;
                int col = bc + wc * 32 + j * 16 + mrow;
                float vv = acc[i][j][e] + bias[col];
                if (relu) vv = fmaxf(vv, 0.f);
                size_t idx = (size_t)row * N + col;
                if (resid) vv += resid[idx];
                if (outF) outF[idx] = vv;
                if (outB) outB[idx] = f2bf(vv);
            }
}

// ---------------- Flash attention (bf16 WMMA, f32 online softmax) ----------------
// grid: (S/64, B*H), block: 128 (4 waves); wave -> 16 query rows
__global__ __launch_bounds__(128)
void attention_kernel(const bf16* __restrict__ q, const bf16* __restrict__ k,
                      const bf16* __restrict__ v, const int* __restrict__ mask,
                      bf16* __restrict__ out) {
    __shared__ bf16 Qs[64][64];       // [q row][d]
    __shared__ bf16 Ks[32][64];       // [key][d]
    __shared__ bf16 Vs[64][32];       // transposed: [d][key]
    __shared__ bf16 Ps[4][16][32];    // per-wave P staging (C-layout -> A-frag)

    const int t    = threadIdx.x;
    const int lane = t & 31;
    const int wave = t >> 5;
    const int b    = blockIdx.y / Hh;
    const int h    = blockIdx.y % Hh;
    const int q0   = blockIdx.x * 64;
    const size_t headoff = (size_t)h * Dk;

    // load Q block (64x64) via async-to-LDS
    {
        int r = t >> 1, c0 = (t & 1) * 32;
        const bf16* gq = &q[((size_t)(b * Ss + q0 + r)) * Ee + headoff + c0];
#pragma unroll
        for (int u = 0; u < 4; ++u) async_b128(gq + 8 * u, &Qs[r][c0 + 8 * u]);
    }
    wait_async0();
    __syncthreads();

    const int mrow = lane & 15;
    const int g    = lane >> 4;

    v16bf aQ[2];
#pragma unroll
    for (int c = 0; c < 2; ++c)
        aQ[c] = cat16(*(const v8bf*)&Qs[wave * 16 + mrow][c * 32 + g * 8],
                      *(const v8bf*)&Qs[wave * 16 + mrow][c * 32 + 16 + g * 8]);

    const v8f zf = {0.f,0.f,0.f,0.f,0.f,0.f,0.f,0.f};
    v8f Oacc[4];
#pragma unroll
    for (int c = 0; c < 4; ++c) Oacc[c] = zf;

    float m_i[8], l_i[8];
#pragma unroll
    for (int e = 0; e < 8; ++e) { m_i[e] = -__builtin_inff(); l_i[e] = 0.f; }

    const float scale = 0.125f;  // 1/sqrt(64)

    for (int kb = 0; kb < Ss / 32; ++kb) {
        __syncthreads();  // previous iteration's LDS reads done
        {
            int r = t >> 2, c0 = (t & 3) * 16;
            const bf16* gk = &k[((size_t)(b * Ss + kb * 32 + r)) * Ee + headoff + c0];
            async_b128(gk,     &Ks[r][c0]);        // K tile: async, contiguous
            async_b128(gk + 8, &Ks[r][c0 + 8]);
            const v8bf* gv = (const v8bf*)&v[((size_t)(b * Ss + kb * 32 + r)) * Ee + headoff + c0];
            v8bf v0 = gv[0], v1 = gv[1];           // V tile: VGPR transpose
#pragma unroll
            for (int u = 0; u < 8; ++u) {
                Vs[c0 + u][r]     = v0[u];
                Vs[c0 + 8 + u][r] = v1[u];
            }
            if (kb + 1 < Ss / 32) {                // gfx1250 global_prefetch_b8
                __builtin_prefetch(&k[((size_t)(b * Ss + (kb + 1) * 32 + r)) * Ee + headoff + c0], 0, 1);
                __builtin_prefetch(&v[((size_t)(b * Ss + (kb + 1) * 32 + r)) * Ee + headoff + c0], 0, 1);
            }
        }
        wait_async0();
        __syncthreads();

        // scores: Q(16x64) @ K^T(64x32) -> two 16x16 C tiles
        v8f sc[2]; sc[0] = zf; sc[1] = zf;
#pragma unroll
        for (int j = 0; j < 2; ++j)
#pragma unroll
            for (int c = 0; c < 2; ++c) {
                int kr = j * 16 + mrow;
                v16bf bK = cat16(*(const v8bf*)&Ks[kr][c * 32 + g * 8],
                                 *(const v8bf*)&Ks[kr][c * 32 + 16 + g * 8]);
                sc[j] = __builtin_amdgcn_wmma_f32_16x16x32_bf16(
                    false, aQ[c], false, bK, (short)0, sc[j], false, false);
            }

        float pv[2][8];
#pragma unroll
        for (int j = 0; j < 2; ++j)
#pragma unroll
            for (int e = 0; e < 8; ++e) {
                float s = sc[j][e] * scale;
                int qg = q0 + wave * 16 + e + 8 * g;
                int kg = kb * 32 + j * 16 + mrow;
                if (mask[(size_t)qg * Ss + kg] == 0) s = -1e9f;
                pv[j][e] = s;
            }

        // online softmax update (row reductions within 16-lane groups)
#pragma unroll
        for (int e = 0; e < 8; ++e) {
            float r = fmaxf(pv[0][e], pv[1][e]);
            for (int off = 1; off < 16; off <<= 1) r = fmaxf(r, __shfl_xor(r, off, 32));
            float mn = fmaxf(m_i[e], r);
            float al = __expf(m_i[e] - mn);
            pv[0][e] = __expf(pv[0][e] - mn);
            pv[1][e] = __expf(pv[1][e] - mn);
            float rs = pv[0][e] + pv[1][e];
            for (int off = 1; off < 16; off <<= 1) rs += __shfl_xor(rs, off, 32);
            l_i[e] = l_i[e] * al + rs;
            m_i[e] = mn;
#pragma unroll
            for (int c = 0; c < 4; ++c) Oacc[c][e] *= al;
        }

        // stage P (C-layout f32 -> A-fragment bf16 via LDS)
#pragma unroll
        for (int j = 0; j < 2; ++j)
#pragma unroll
            for (int e = 0; e < 8; ++e)
                Ps[wave][e + 8 * g][j * 16 + mrow] = f2bf(pv[j][e]);
        __syncthreads();

        v16bf aP = cat16(*(const v8bf*)&Ps[wave][mrow][g * 8],
                         *(const v8bf*)&Ps[wave][mrow][16 + g * 8]);
#pragma unroll
        for (int c = 0; c < 4; ++c) {
            v16bf bV = cat16(*(const v8bf*)&Vs[c * 16 + mrow][g * 8],
                             *(const v8bf*)&Vs[c * 16 + mrow][16 + g * 8]);
            Oacc[c] = __builtin_amdgcn_wmma_f32_16x16x32_bf16(
                false, aP, false, bV, (short)0, Oacc[c], false, false);
        }
    }

    // final normalize + store
#pragma unroll
    for (int c = 0; c < 4; ++c)
#pragma unroll
        for (int e = 0; e < 8; ++e) {
            int row = q0 + wave * 16 + e + 8 * g;
            int d   = c * 16 + mrow;
            out[((size_t)(b * Ss + row)) * Ee + headoff + d] =
                f2bf(Oacc[c][e] * (1.f / l_i[e]));
        }
}

// ---------------- host orchestration ----------------
extern "C" void kernel_launch(void* const* d_in, const int* in_sizes, int n_in,
                              void* d_out, int out_size, void* d_ws, size_t ws_size,
                              hipStream_t stream) {
    (void)in_sizes; (void)n_in; (void)out_size; (void)ws_size;

    const float* x    = (const float*)d_in[0];
    const int*   mask = (const int*)  d_in[1];
    const float* wq = (const float*)d_in[2];  const float* bq = (const float*)d_in[3];
    const float* wk = (const float*)d_in[4];  const float* bk = (const float*)d_in[5];
    const float* wv = (const float*)d_in[6];  const float* bv = (const float*)d_in[7];
    const float* wo = (const float*)d_in[8];  const float* bo = (const float*)d_in[9];
    const float* w1 = (const float*)d_in[10]; const float* b1 = (const float*)d_in[11];
    const float* w2 = (const float*)d_in[12]; const float* b2 = (const float*)d_in[13];
    const float* ln1a = (const float*)d_in[14]; const float* ln1b = (const float*)d_in[15];
    const float* ln2a = (const float*)d_in[16]; const float* ln2b = (const float*)d_in[17];
    float* out = (float*)d_out;

    char* ws = (char*)d_ws;
    const size_t MB = (size_t)1 << 20;
    bf16* wq_bf = (bf16*)(ws +   0 * MB);   // 2 MB each
    bf16* wk_bf = (bf16*)(ws +   2 * MB);
    bf16* wv_bf = (bf16*)(ws +   4 * MB);
    bf16* wo_bf = (bf16*)(ws +   6 * MB);
    bf16* w1_bf = (bf16*)(ws +   8 * MB);   // 8 MB
    bf16* w2_bf = (bf16*)(ws +  16 * MB);   // 8 MB
    bf16* n1_bf = (bf16*)(ws +  24 * MB);   // 16 MB
    bf16* q_bf  = (bf16*)(ws +  40 * MB);   // 16 MB
    bf16* k_bf  = (bf16*)(ws +  56 * MB);   // 16 MB
    bf16* v_bf  = (bf16*)(ws +  72 * MB);   // 16 MB
    bf16* at_bf = n1_bf;                    // reuse: n1 dead after QKV GEMMs
    float* h1   = (float*)(ws + 88 * MB);   // 32 MB
    bf16* n2_bf = q_bf;                     // reuse: q dead after attention
    bf16* f1_bf = (bf16*)(ws + 120 * MB);   // 64 MB

    const int EE   = Ee * Ee;    // 1048576
    const int EDFF = Ee * Dff;   // 4194304

    // weights -> bf16
    f32_to_bf16_kernel<<<EE / 256, 256, 0, stream>>>(wq, wq_bf, EE);
    f32_to_bf16_kernel<<<EE / 256, 256, 0, stream>>>(wk, wk_bf, EE);
    f32_to_bf16_kernel<<<EE / 256, 256, 0, stream>>>(wv, wv_bf, EE);
    f32_to_bf16_kernel<<<EE / 256, 256, 0, stream>>>(wo, wo_bf, EE);
    f32_to_bf16_kernel<<<EDFF / 256, 256, 0, stream>>>(w1, w1_bf, EDFF);
    f32_to_bf16_kernel<<<EDFF / 256, 256, 0, stream>>>(w2, w2_bf, EDFF);

    // LN1
    layernorm_bf16_kernel<<<Mm, 256, 0, stream>>>(x, ln1a, ln1b, n1_bf);

    // Q/K/V projections
    dim3 gE(Ee / 128, Mm / 128);
    gemm_bf16_kernel<<<gE, 256, 0, stream>>>(n1_bf, wq_bf, bq, nullptr, q_bf, nullptr, Mm, Ee, Ee, 0);
    gemm_bf16_kernel<<<gE, 256, 0, stream>>>(n1_bf, wk_bf, bk, nullptr, k_bf, nullptr, Mm, Ee, Ee, 0);
    gemm_bf16_kernel<<<gE, 256, 0, stream>>>(n1_bf, wv_bf, bv, nullptr, v_bf, nullptr, Mm, Ee, Ee, 0);

    // attention
    attention_kernel<<<dim3(Ss / 64, Bb * Hh), 128, 0, stream>>>(q_bf, k_bf, v_bf, mask, at_bf);

    // out projection + residual(x) -> h1 (f32)
    gemm_bf16_kernel<<<gE, 256, 0, stream>>>(at_bf, wo_bf, bo, h1, nullptr, x, Mm, Ee, Ee, 0);

    // LN2
    layernorm_bf16_kernel<<<Mm, 256, 0, stream>>>(h1, ln2a, ln2b, n2_bf);

    // FFN1 (+ReLU)
    dim3 gF(Dff / 128, Mm / 128);
    gemm_bf16_kernel<<<gF, 256, 0, stream>>>(n2_bf, w1_bf, b1, nullptr, f1_bf, nullptr, Mm, Dff, Ee, 1);

    // FFN2 + residual(h1) -> out
    gemm_bf16_kernel<<<gE, 256, 0, stream>>>(f1_bf, w2_bf, b2, out, nullptr, h1, Mm, Ee, Dff, 0);
}
